// GATNet_41120016892605
// MI455X (gfx1250) — matrix operand order repaired and checked by reading, not compiled, not run on hardware
//
#include <hip/hip_runtime.h>
#include <hip/hip_bf16.h>

typedef __attribute__((ext_vector_type(16))) _Float16 v16h;
typedef __attribute__((ext_vector_type(8)))  float    v8f;

#define F_OUT 64           // both layers produce 64 features
#define TILE_M 128
#define KSTEP  64          // K-chunk per staged tile (2 WMMA k-subtiles)
#define LDA_H  72          // padded row stride (halfs) for A tile in LDS
#define LDB_H  72          // padded col stride (halfs) for B^T tile in LDS

union V16HU { unsigned int u[8]; v16h v; };

// ---------------------------------------------------------------------------
// float atomic max via ordered-int trick (works with -inf init, mixed signs)
// ---------------------------------------------------------------------------
__device__ inline void atomicMaxF(float* addr, float val) {
    if (val >= 0.0f) {
        atomicMax(reinterpret_cast<int*>(addr), __float_as_int(val));
    } else {
        atomicMin(reinterpret_cast<unsigned int*>(addr), __float_as_uint(val));
    }
}

// ---------------------------------------------------------------------------
// Init: zero aggregation buffers / output, set segment-max buffers to -inf
// ---------------------------------------------------------------------------
__global__ void gat_init_kernel(float* m1, float* den1, float* agg1,
                                float* m2, float* den2, float* out, int n) {
    int i = blockIdx.x * blockDim.x + threadIdx.x;
    const float ninf = -__builtin_inff();
    if (i < n * 64) { out[i] = 0.0f; agg1[i] = 0.0f; }
    if (i < n * 8)  { m1[i] = ninf;  den1[i] = 0.0f; }
    if (i < n)      { m2[i] = ninf;  den2[i] = 0.0f; }
}

// ---------------------------------------------------------------------------
// WMMA GEMM: C[M x 64] = A[M x K] (fp32) @ B[K x 64] (fp32), f16 in, f32 acc.
// 256 threads = 8 waves; block tile 128x64; wave tile 16x64; K chunk 64.
// Row guard is a clamp (no divergence); staging uses b128 loads.
// Requires K % 64 == 0 (512 and 64 here).
// ---------------------------------------------------------------------------
__global__ __launch_bounds__(256)
void gat_gemm_wmma_kernel(const float* __restrict__ A,
                          const float* __restrict__ B,
                          float* __restrict__ C, int M, int K) {
    __shared__ _Float16 sA[TILE_M * LDA_H];   // 128 rows x 64 k (padded)
    __shared__ _Float16 sBt[F_OUT * LDB_H];   // 64 cols x 64 k (padded)

    const int tid  = threadIdx.x;
    const int wave = tid >> 5;
    const int lane = tid & 31;
    const int r0   = blockIdx.x * TILE_M;
    const int mrow = wave * 16;              // wave's 16-row strip inside tile

    v8f acc[4];
    #pragma unroll
    for (int n = 0; n < 4; ++n) acc[n] = (v8f){0,0,0,0,0,0,0,0};

    for (int kc = 0; kc < K; kc += KSTEP) {
        // --- stage A tile (128x64 fp32 -> f16) with float4 loads ---
        // rows clamped to M-1: garbage rows feed only never-stored C rows.
        #pragma unroll
        for (int i = 0; i < 8; ++i) {
            int e  = tid + i * 256;          // 0..2047 float4 slots
            int r  = e >> 4;                 // 16 float4 per row
            int k4 = (e & 15) * 4;
            int rg = r0 + r; rg = (rg < M) ? rg : (M - 1);
            float4 v = *reinterpret_cast<const float4*>(
                &A[(size_t)rg * K + kc + k4]);
            _Float16* p = &sA[r * LDA_H + k4];
            p[0] = (_Float16)v.x; p[1] = (_Float16)v.y;
            p[2] = (_Float16)v.z; p[3] = (_Float16)v.w;
        }
        // --- stage B tile transposed (64x64 fp32 -> f16, K-contig per col) ---
        #pragma unroll
        for (int i = 0; i < 4; ++i) {
            int e  = tid + i * 256;          // 0..1023 float4 slots
            int k  = e >> 4;                 // B row (K dim)
            int c4 = (e & 15) * 4;
            float4 v = *reinterpret_cast<const float4*>(
                &B[(size_t)(kc + k) * F_OUT + c4]);
            sBt[(c4 + 0) * LDB_H + k] = (_Float16)v.x;
            sBt[(c4 + 1) * LDB_H + k] = (_Float16)v.y;
            sBt[(c4 + 2) * LDB_H + k] = (_Float16)v.z;
            sBt[(c4 + 3) * LDB_H + k] = (_Float16)v.w;
        }
        __syncthreads();

        const unsigned int* aRow =
            reinterpret_cast<const unsigned int*>(&sA[(mrow + (lane & 15)) * LDA_H]);
        const int kb2 = (lane >> 4) * 4;     // A dword-pair offset (0 or 4)
        const int kb8 = (lane >> 4) * 8;     // unused; B loads use kb2 like A

        #pragma unroll
        for (int ks = 0; ks < KSTEP; ks += 32) {
            const int ks2 = ks >> 1;         // dword offset of k-subtile
            // A fragment: 16x32 f16, lane = M (mod 16), hi-half lanes K+8
            V16HU af;
            #pragma unroll
            for (int j = 0; j < 4; ++j) af.u[j]     = aRow[ks2 + kb2 + j];
            #pragma unroll
            for (int j = 0; j < 4; ++j) af.u[4 + j] = aRow[ks2 + 8 + kb2 + j];

            // B fragments: 32x16 f16 per n-tile, lane = N col, same k split as A
            #pragma unroll
            for (int n = 0; n < 4; ++n) {
                const unsigned int* bCol = reinterpret_cast<const unsigned int*>(
                    &sBt[(n * 16 + (lane & 15)) * LDB_H]);
                V16HU bf;
                #pragma unroll
                for (int j = 0; j < 4; ++j) bf.u[j] = bCol[ks2 + kb2 + j];
                #pragma unroll
                for (int j = 0; j < 4; ++j) bf.u[4 + j] = bCol[ks2 + 8 + kb2 + j];
                acc[n] = __builtin_amdgcn_wmma_f32_16x16x32_f16(
                    false, af.v, false, bf.v, (short)0, acc[n], false, false);
            }
        }
        __syncthreads();
    }

    // --- epilogue: C layout lanes 0-15 -> M=v, lanes 16-31 -> M=v+8 ---
    #pragma unroll
    for (int n = 0; n < 4; ++n) {
        #pragma unroll
        for (int v = 0; v < 8; ++v) {
            int row = mrow + ((lane < 16) ? v : v + 8);
            int col = n * 16 + (lane & 15);
            int gr = r0 + row;
            if (gr < M) C[(size_t)gr * F_OUT + col] = acc[n][v];
        }
    }
}

// ---------------------------------------------------------------------------
// Per-node attention scores: es[n,h] = <h[n,h,:], a_src[h,:]>, same for ed.
// ---------------------------------------------------------------------------
__global__ void gat_scores_kernel(const float* __restrict__ h,
                                  const float* __restrict__ a_src,
                                  const float* __restrict__ a_dst,
                                  float* __restrict__ es, float* __restrict__ ed,
                                  int n, int H, int Fh) {
    int i = blockIdx.x * blockDim.x + threadIdx.x;
    if (i >= n * H) return;
    int node = i / H, head = i - node * H;
    const float* row = h + (size_t)node * F_OUT + head * Fh;
    const float* as  = a_src + head * Fh;
    const float* ad  = a_dst + head * Fh;
    float s = 0.0f, d = 0.0f;
    for (int f = 0; f < Fh; ++f) { s += row[f] * as[f]; d += row[f] * ad[f]; }
    es[i] = s; ed[i] = d;
}

__device__ inline void edge_ids(const int* __restrict__ src,
                                const int* __restrict__ dst,
                                int e, int E, int& s, int& d) {
    if (e < E) { s = src[e]; d = dst[e]; }
    else       { s = e - E;  d = e - E; }      // appended self-loops
}

// ---------------------------------------------------------------------------
// Pass 1: segment max of leaky-relu edge scores (per dst, per head)
// ---------------------------------------------------------------------------
__global__ void gat_edge_max_kernel(const int* __restrict__ src,
                                    const int* __restrict__ dst,
                                    const float* __restrict__ es,
                                    const float* __restrict__ ed,
                                    float* __restrict__ m,
                                    int E, int n, int H) {
    int e = blockIdx.x * blockDim.x + threadIdx.x;
    if (e >= E + n) return;
    int s, d; edge_ids(src, dst, e, E, s, d);
    for (int h = 0; h < H; ++h) {
        float sc = es[s * H + h] + ed[d * H + h];
        sc = (sc > 0.0f) ? sc : 0.2f * sc;
        atomicMaxF(&m[d * H + h], sc);
    }
}

// ---------------------------------------------------------------------------
// Pass 2: segment sum of exp(score - max)
// ---------------------------------------------------------------------------
__global__ void gat_edge_sum_kernel(const int* __restrict__ src,
                                    const int* __restrict__ dst,
                                    const float* __restrict__ es,
                                    const float* __restrict__ ed,
                                    const float* __restrict__ m,
                                    float* __restrict__ den,
                                    int E, int n, int H) {
    int e = blockIdx.x * blockDim.x + threadIdx.x;
    if (e >= E + n) return;
    int s, d; edge_ids(src, dst, e, E, s, d);
    for (int h = 0; h < H; ++h) {
        float sc = es[s * H + h] + ed[d * H + h];
        sc = (sc > 0.0f) ? sc : 0.2f * sc;
        atomicAdd(&den[d * H + h], __expf(sc - m[d * H + h]));
    }
}

// ---------------------------------------------------------------------------
// Pass 3: weighted aggregation out[dst] += alpha * h[src]
// ---------------------------------------------------------------------------
__global__ void gat_edge_agg_kernel(const int* __restrict__ src,
                                    const int* __restrict__ dst,
                                    const float* __restrict__ es,
                                    const float* __restrict__ ed,
                                    const float* __restrict__ m,
                                    const float* __restrict__ den,
                                    const float* __restrict__ hfeat,
                                    float* __restrict__ out,
                                    int E, int n, int H, int Fh) {
    int e = blockIdx.x * blockDim.x + threadIdx.x;
    if (e >= E + n) return;
    int s, d; edge_ids(src, dst, e, E, s, d);
    const float* hr = hfeat + (size_t)s * F_OUT;
    float* orow = out + (size_t)d * F_OUT;
    for (int h = 0; h < H; ++h) {
        float sc = es[s * H + h] + ed[d * H + h];
        sc = (sc > 0.0f) ? sc : 0.2f * sc;
        float alpha = __expf(sc - m[d * H + h]) / den[d * H + h];
        for (int f = 0; f < Fh; ++f) {
            atomicAdd(&orow[h * Fh + f], alpha * hr[h * Fh + f]);
        }
    }
}

// ---------------------------------------------------------------------------
// x1 = elu(agg + b)  (in place)
// ---------------------------------------------------------------------------
__global__ void gat_bias_elu_kernel(float* __restrict__ x,
                                    const float* __restrict__ b, int n) {
    int i = blockIdx.x * blockDim.x + threadIdx.x;
    if (i >= n * F_OUT) return;
    float v = x[i] + b[i & (F_OUT - 1)];
    x[i] = (v > 0.0f) ? v : (__expf(v) - 1.0f);
}

// ---------------------------------------------------------------------------
// out = log_softmax(out + b2), one wave32 per node (64 classes = 2/lane)
// ---------------------------------------------------------------------------
__global__ void gat_logsoftmax_kernel(float* __restrict__ out,
                                      const float* __restrict__ b, int n) {
    int warp = (blockIdx.x * blockDim.x + threadIdx.x) >> 5;
    int lane = threadIdx.x & 31;
    if (warp >= n) return;
    float a0 = out[(size_t)warp * 64 + lane]      + b[lane];
    float a1 = out[(size_t)warp * 64 + 32 + lane] + b[32 + lane];
    float mx = fmaxf(a0, a1);
    #pragma unroll
    for (int o = 16; o > 0; o >>= 1) mx = fmaxf(mx, __shfl_xor(mx, o, 32));
    float sum = __expf(a0 - mx) + __expf(a1 - mx);
    #pragma unroll
    for (int o = 16; o > 0; o >>= 1) sum += __shfl_xor(sum, o, 32);
    float l = mx + logf(sum);
    out[(size_t)warp * 64 + lane]      = a0 - l;
    out[(size_t)warp * 64 + 32 + lane] = a1 - l;
}

// ---------------------------------------------------------------------------
extern "C" void kernel_launch(void* const* d_in, const int* in_sizes, int n_in,
                              void* d_out, int out_size, void* d_ws, size_t ws_size,
                              hipStream_t stream) {
    const float* x        = (const float*)d_in[0];
    const int*   ei       = (const int*)  d_in[1];
    const float* W1       = (const float*)d_in[2];
    const float* att1_src = (const float*)d_in[3];
    const float* att1_dst = (const float*)d_in[4];
    const float* b1       = (const float*)d_in[5];
    const float* W2       = (const float*)d_in[6];
    const float* att2_src = (const float*)d_in[7];
    const float* att2_dst = (const float*)d_in[8];
    const float* b2       = (const float*)d_in[9];

    const int FIN = 512;
    const int n   = in_sizes[0] / FIN;     // 100000 nodes
    const int E   = in_sizes[1] / 2;       // 3.2M edges (self-loops appended)
    const int* src = ei;
    const int* dst = ei + E;

    float* out = (float*)d_out;

    // workspace carve-up (floats)
    float* ws  = (float*)d_ws;
    float* h1   = ws;                      // n*64
    float* agg1 = h1   + (size_t)n * 64;   // n*64 (becomes x1 after bias+elu)
    float* h2   = agg1 + (size_t)n * 64;   // n*64
    float* es1  = h2   + (size_t)n * 64;   // n*8
    float* ed1  = es1  + (size_t)n * 8;
    float* m1   = ed1  + (size_t)n * 8;
    float* den1 = m1   + (size_t)n * 8;
    float* es2  = den1 + (size_t)n * 8;    // n*1
    float* ed2  = es2  + (size_t)n;
    float* m2   = ed2  + (size_t)n;
    float* den2 = m2   + (size_t)n;

    const int ET = E + n;
    dim3 blk(256);
    dim3 gInit((n * 64 + 255) / 256);
    dim3 gGemm((n + TILE_M - 1) / TILE_M);
    dim3 gEdge((ET + 255) / 256);
    dim3 gSc8((n * 8 + 255) / 256);
    dim3 gSc1((n + 255) / 256);
    dim3 gElu((n * 64 + 255) / 256);
    dim3 gLsm(((size_t)n * 32 + 255) / 256);

    gat_init_kernel<<<gInit, blk, 0, stream>>>(m1, den1, agg1, m2, den2, out, n);

    // ----- layer 1 -----
    gat_gemm_wmma_kernel<<<gGemm, blk, 0, stream>>>(x, W1, h1, n, FIN);
    gat_scores_kernel<<<gSc8, blk, 0, stream>>>(h1, att1_src, att1_dst, es1, ed1, n, 8, 8);
    gat_edge_max_kernel<<<gEdge, blk, 0, stream>>>(src, dst, es1, ed1, m1, E, n, 8);
    gat_edge_sum_kernel<<<gEdge, blk, 0, stream>>>(src, dst, es1, ed1, m1, den1, E, n, 8);
    gat_edge_agg_kernel<<<gEdge, blk, 0, stream>>>(src, dst, es1, ed1, m1, den1, h1, agg1, E, n, 8, 8);
    gat_bias_elu_kernel<<<gElu, blk, 0, stream>>>(agg1, b1, n);   // agg1 -> x1

    // ----- layer 2 -----
    gat_gemm_wmma_kernel<<<gGemm, blk, 0, stream>>>(agg1, W2, h2, n, 64);
    gat_scores_kernel<<<gSc1, blk, 0, stream>>>(h2, att2_src, att2_dst, es2, ed2, n, 1, 64);
    gat_edge_max_kernel<<<gEdge, blk, 0, stream>>>(src, dst, es2, ed2, m2, E, n, 1);
    gat_edge_sum_kernel<<<gEdge, blk, 0, stream>>>(src, dst, es2, ed2, m2, den2, E, n, 1);
    gat_edge_agg_kernel<<<gEdge, blk, 0, stream>>>(src, dst, es2, ed2, m2, den2, h2, out, E, n, 1, 64);
    gat_logsoftmax_kernel<<<gLsm, blk, 0, stream>>>(out, b2, n);
}
